// BoringAttention_73658689126761
// MI455X (gfx1250) — matrix operand order repaired and checked by reading, not compiled, not run on hardware
//
#include <hip/hip_runtime.h>
#include <hip/hip_bf16.h>
#include <cstdint>

typedef __attribute__((ext_vector_type(16))) _Float16 v16h;
typedef __attribute__((ext_vector_type(8)))  _Float16 v8h;
typedef __attribute__((ext_vector_type(8)))  float    v8f;
typedef _Float16 half_t;
typedef __attribute__((address_space(3))) half_t lds_half_t;

#define D_MODEL 1024
#define N_HEADS 16
#define DIM_HEAD 64
#define INNER 1024
#define BATCH 2
#define SEQ 2048
#define NMEM 16
#define JTOT (NMEM + SEQ)      // 2064
#define JP 2080                // padded to multiple of 32
#define MROWS (BATCH * SEQ)    // 4096

// ---------------- workspace layout (in _Float16 units) ----------------
constexpr size_t SZ_XH  = (size_t)MROWS * D_MODEL;
constexpr size_t SZ_WQH = (size_t)D_MODEL * INNER;
constexpr size_t SZ_WKVH= (size_t)D_MODEL * 2 * INNER;
constexpr size_t SZ_WOH = (size_t)INNER * D_MODEL;
constexpr size_t SZ_Q   = (size_t)BATCH * N_HEADS * SEQ * DIM_HEAD;   // scaled q [b][h][n][d]
constexpr size_t SZ_K   = (size_t)BATCH * N_HEADS * DIM_HEAD * JP;    // K TRANSPOSED [b][h][d][j]
constexpr size_t SZ_V   = (size_t)BATCH * N_HEADS * JP * DIM_HEAD;    // V row-major [b][h][j][d]
constexpr size_t SZ_AO  = (size_t)MROWS * INNER;

constexpr size_t OFF_XH  = 0;
constexpr size_t OFF_WQH = OFF_XH  + SZ_XH;
constexpr size_t OFF_WKVH= OFF_WQH + SZ_WQH;
constexpr size_t OFF_WOH = OFF_WKVH+ SZ_WKVH;
constexpr size_t OFF_Q   = OFF_WOH + SZ_WOH;
constexpr size_t OFF_K   = OFF_Q   + SZ_Q;
constexpr size_t OFF_V   = OFF_K   + SZ_K;
constexpr size_t OFF_AO  = OFF_V   + SZ_V;

// ---------------- CDNA5 async copy (global -> LDS, ASYNCcnt) ----------------
__device__ inline uint32_t lds_addr32(void* p) {
  return (uint32_t)(uintptr_t)(lds_half_t*)(half_t*)p;
}
__device__ inline void async_ld16(void* lds_p, const half_t* g) {
  uint32_t l = lds_addr32(lds_p);
  asm volatile("global_load_async_to_lds_b128 %0, %1, off" :: "v"(l), "v"(g) : "memory");
}
__device__ inline void wait_async0() {
#if __has_builtin(__builtin_amdgcn_s_wait_asynccnt)
  __builtin_amdgcn_s_wait_asynccnt(0);
#else
  asm volatile("s_wait_asynccnt 0x0" ::: "memory");
#endif
}
__device__ inline void wait_async6() {   // <=6 outstanding: previous tile (6 ops/thread) done
#if __has_builtin(__builtin_amdgcn_s_wait_asynccnt)
  __builtin_amdgcn_s_wait_asynccnt(6);
#else
  asm volatile("s_wait_asynccnt 0x6" ::: "memory");
#endif
}

// ---------------- WMMA fragment helpers (ISA 7.12.2, wave32) ----------------
__device__ inline v16h cat8(v8h lo, v8h hi) {
  return __builtin_shufflevector(lo, hi, 0, 1, 2, 3, 4, 5, 6, 7, 8, 9, 10, 11, 12, 13, 14, 15);
}

// A (16x32 f16): lanes 0-15: M=lane, K in {0..7,16..23}; lanes 16-31: M=lane-16, K in {8..15,24..31}
__device__ inline v16h load_a_frag(const half_t* p, int ld, int lane) {
  int m  = lane & 15;
  int kb = (lane >> 4) * 8;
  const half_t* row = p + (size_t)m * ld + kb;
  return cat8(*(const v8h*)row, *(const v8h*)(row + 16));
}

// B (32x16 f16): lanes 0-15: N=lane, K in {0..7,16..23}; lanes 16-31: N=lane-16, K in {8..15,24..31}. Source row-major KxN.
__device__ inline v16h load_b_frag(const half_t* p, int ld, int lane) {
  v16h b;
  int n  = lane & 15;
  int kb = (lane >> 4) * 8;
#pragma unroll
  for (int s = 0; s < 8; ++s) {
    b[s]     = p[(size_t)(kb + s) * ld + n];
    b[8 + s] = p[(size_t)(16 + kb + s) * ld + n];
  }
  return b;
}

// B for the head-mix: B[c][n] = S[(i*16+c)*32 + jh*16 + n] for c<16, else 0 (K padded 16->32); N=lane&15, K in table order
__device__ inline v16h load_mixB_frag(const half_t* base, int lane) {
  v16h b;
  int n  = lane & 15;
  int kb = (lane >> 4) * 8;
#pragma unroll
  for (int s = 0; s < 8; ++s) {
    b[s]     = base[(size_t)(kb + s) * 32 + n];
    b[8 + s] = (half_t)0.0f;
  }
  return b;
}

// A for the head-mix: A[m=k][c=h] = M[h*16+k] for h<16, else 0 (zero-padded K=32)
__device__ inline v16h load_mixA_frag(const float* M, int lane) {
  v16h a;
  int m  = lane & 15;
  int kb = (lane >> 4) * 8;
#pragma unroll
  for (int r = 0; r < 8; ++r) {
#pragma unroll
    for (int pos = 0; pos < 2; ++pos) {
      int c = (r < 4) ? (kb + 2 * r + pos) : (16 + kb + 2 * (r - 4) + pos);
      a[2 * r + pos] = (c < 16) ? (half_t)M[c * 16 + m] : (half_t)0.0f;
    }
  }
  return a;
}

__device__ inline v8f wmma_f16(v16h a, v16h b, v8f c) {
  return __builtin_amdgcn_wmma_f32_16x16x32_f16(false, a, false, b, (short)0, c, false, false);
}

// ---------------- phase 0: casts + mem-kv placement + pad zeroing ----------------
__global__ __launch_bounds__(256) void prep_kernel(const float* __restrict__ x,
                                                   const float* __restrict__ Wq,
                                                   const float* __restrict__ Wkv,
                                                   const float* __restrict__ Wo,
                                                   const float* __restrict__ mem_k,
                                                   const float* __restrict__ mem_v,
                                                   half_t* __restrict__ ws) {
  constexpr size_t SZ_MEM = (size_t)BATCH * N_HEADS * NMEM * DIM_HEAD;
  constexpr size_t SZ_PAD = (size_t)BATCH * N_HEADS * (JP - JTOT) * DIM_HEAD;
  constexpr size_t n0 = SZ_XH;
  constexpr size_t n1 = n0 + SZ_WQH;
  constexpr size_t n2 = n1 + SZ_WKVH;
  constexpr size_t n3 = n2 + SZ_WOH;
  constexpr size_t n4 = n3 + SZ_MEM;   // mem_k -> Kt
  constexpr size_t n5 = n4 + SZ_MEM;   // mem_v -> V
  constexpr size_t n6 = n5 + SZ_PAD;   // zero Kt pad cols
  constexpr size_t n7 = n6 + SZ_PAD;   // zero V pad rows
  size_t tid = (size_t)blockIdx.x * blockDim.x + threadIdx.x;
  size_t stride = (size_t)gridDim.x * blockDim.x;
  for (size_t i = tid; i < n7; i += stride) {
    if (i < n0)       ws[OFF_XH + i] = (half_t)x[i];
    else if (i < n1)  ws[OFF_WQH + (i - n0)] = (half_t)Wq[i - n0];
    else if (i < n2)  ws[OFF_WKVH + (i - n1)] = (half_t)Wkv[i - n1];
    else if (i < n3)  ws[OFF_WOH + (i - n2)] = (half_t)Wo[i - n2];
    else if (i < n5) {
      int is_v = i >= n4;
      size_t idx = i - (is_v ? n4 : n3);
      int b = (int)(idx / (N_HEADS * NMEM * DIM_HEAD));
      size_t r = idx % (N_HEADS * NMEM * DIM_HEAD);
      int h = (int)(r / (NMEM * DIM_HEAD));
      int j = (int)((r / DIM_HEAD) % NMEM);
      int d = (int)(r % DIM_HEAD);
      float val = (is_v ? mem_v : mem_k)[((size_t)h * NMEM + j) * DIM_HEAD + d];
      if (is_v)
        ws[OFF_V + ((((size_t)b * N_HEADS + h) * JP) + j) * DIM_HEAD + d] = (half_t)val;
      else  // transposed K: [b][h][d][j]
        ws[OFF_K + ((((size_t)b * N_HEADS + h) * DIM_HEAD) + d) * JP + j] = (half_t)val;
    } else {
      int is_v = i >= n6;
      size_t idx = i - (is_v ? n6 : n5);
      int b = (int)(idx / (N_HEADS * (JP - JTOT) * DIM_HEAD));
      size_t r = idx % (N_HEADS * (JP - JTOT) * DIM_HEAD);
      int h = (int)(r / ((JP - JTOT) * DIM_HEAD));
      int j = JTOT + (int)((r / DIM_HEAD) % (JP - JTOT));
      int d = (int)(r % DIM_HEAD);
      if (is_v)
        ws[OFF_V + ((((size_t)b * N_HEADS + h) * JP) + j) * DIM_HEAD + d] = (half_t)0.0f;
      else
        ws[OFF_K + ((((size_t)b * N_HEADS + h) * DIM_HEAD) + d) * JP + j] = (half_t)0.0f;
    }
  }
}

// ---------------- double-buffered async-staged WMMA GEMM core ----------------
#define BM 64
#define BN 128
#define BK 64
#define LDA_S 72               // 64 + 8 pad (multiple of 8 halves)
#define LDB_S 136              // 128 + 8 pad
#define NT_QKV (D_MODEL / BK)  // 16

// stage one BMxBK A tile and BKxBN B tile into LDS buffers via async DMA (6 ops/thread)
__device__ inline void stage_tile(half_t* Asb, half_t* Bsb,
                                  const half_t* Ag, int lda,
                                  const half_t* Bg, int ldb,
                                  int k0, int tid) {
  int arow = tid >> 2, acg = (tid & 3) * 16;
  const half_t* asrc = Ag + (size_t)arow * lda + k0 + acg;
  async_ld16(Asb + arow * LDA_S + acg,     asrc);
  async_ld16(Asb + arow * LDA_S + acg + 8, asrc + 8);
#pragma unroll
  for (int c = 0; c < 4; ++c) {
    int id = tid + c * 256;
    int row = id >> 4, cg = (id & 15) * 8;
    async_ld16(Bsb + row * LDB_S + cg, Bg + (size_t)(k0 + row) * ldb + cg);
  }
}

// ---------------- phase 1: fused QKV projection GEMM ----------------
__global__ __launch_bounds__(256) void qkv_gemm_kernel(half_t* __restrict__ ws) {
  __shared__ half_t As[2][BM * LDA_S];
  __shared__ half_t Bs[2][BK * LDB_S];
  const half_t* Xh = ws + OFF_XH;
  int tid = threadIdx.x, lane = tid & 31, w = tid >> 5;
  int m0 = blockIdx.y * BM;
  int c0 = blockIdx.x * BN;
  const half_t* Wsrc; int ldw, csrc;
  if (c0 < INNER) { Wsrc = ws + OFF_WQH;  ldw = INNER;     csrc = c0; }
  else            { Wsrc = ws + OFF_WKVH; ldw = 2 * INNER; csrc = c0 - INNER; }
  const half_t* Ag = Xh + (size_t)m0 * D_MODEL;
  const half_t* Bg = Wsrc + csrc;
  int wm = w >> 2, wn = w & 3;  // 2 (M) x 4 (N) waves; each wave 32x32
  v8f acc[2][2] = {};

  stage_tile(As[0], Bs[0], Ag, D_MODEL, Bg, ldw, 0, tid);
  for (int t = 0; t < NT_QKV; ++t) {
    if (t + 1 < NT_QKV) {
      stage_tile(As[(t + 1) & 1], Bs[(t + 1) & 1], Ag, D_MODEL, Bg, ldw, (t + 1) * BK, tid);
      wait_async6();            // tile t done; tile t+1 still in flight during compute
    } else {
      wait_async0();
    }
    __syncthreads();
    const half_t* At = As[t & 1];
    const half_t* Bt = Bs[t & 1];
#pragma unroll
    for (int kc = 0; kc < 2; ++kc) {
      v16h af[2], bf[2];
#pragma unroll
      for (int f = 0; f < 2; ++f) {
        af[f] = load_a_frag(At + (wm * 32 + f * 16) * LDA_S + kc * 32, LDA_S, lane);
        bf[f] = load_b_frag(Bt + (size_t)(kc * 32) * LDB_S + wn * 32 + f * 16, LDB_S, lane);
      }
#pragma unroll
      for (int fm = 0; fm < 2; ++fm)
#pragma unroll
        for (int fn = 0; fn < 2; ++fn)
          acc[fm][fn] = wmma_f16(af[fm], bf[fn], acc[fm][fn]);
    }
    __syncthreads();
  }
  // scatter epilogue: fold softmax scale into Q; K stored transposed
  int nlo = lane & 15, hi = (lane >> 4) * 8;
#pragma unroll
  for (int fm = 0; fm < 2; ++fm)
#pragma unroll
    for (int fn = 0; fn < 2; ++fn)
#pragma unroll
      for (int r = 0; r < 8; ++r) {
        int mg = m0 + wm * 32 + fm * 16 + hi + r;
        int cg = c0 + wn * 32 + fn * 16 + nlo;
        float v = acc[fm][fn][r];
        int b = mg >> 11, n = mg & (SEQ - 1);
        if (cg < INNER) {
          int h = cg >> 6, d = cg & 63;
          ws[OFF_Q + ((((size_t)b * N_HEADS + h) * SEQ) + n) * DIM_HEAD + d] =
              (half_t)(v * 0.125f);   // DIM_HEAD^-0.5 folded into Q
        } else if (cg < 2 * INNER) {
          int cc = cg - INNER, h = cc >> 6, d = cc & 63;
          ws[OFF_K + ((((size_t)b * N_HEADS + h) * DIM_HEAD) + d) * JP + NMEM + n] = (half_t)v;
        } else {
          int cc = cg - 2 * INNER, h = cc >> 6, d = cc & 63;
          ws[OFF_V + ((((size_t)b * N_HEADS + h) * JP) + NMEM + n) * DIM_HEAD + d] = (half_t)v;
        }
      }
}

// ---------------- phase 2: two-pass talking-heads flash attention ----------------
__global__ __launch_bounds__(256) void attn_kernel(const float* __restrict__ pre,
                                                   const float* __restrict__ post,
                                                   half_t* __restrict__ ws) {
  __shared__ half_t bufA[16 * 16 * 32];  // [i][h][j] f16 : raw dots S, later P
  __shared__ float  bufB[16 * 16 * 32];  // [k][i][j] f32 : mixed dots S'; aliased f16 for P'
  half_t* bufBh = reinterpret_cast<half_t*>(bufB);

  int tid = threadIdx.x, lane = tid & 31, w = tid >> 5;
  int b = blockIdx.y;
  int i0 = blockIdx.x * 16;
  int h0 = w * 2;
  const half_t* Qp = ws + OFF_Q;
  const half_t* Kp = ws + OFF_K;   // transposed [b][h][d][j]
  const half_t* Vp = ws + OFF_V;

  v16h qf[2][2];
#pragma unroll
  for (int hh = 0; hh < 2; ++hh)
#pragma unroll
    for (int kc = 0; kc < 2; ++kc)
      qf[hh][kc] = load_a_frag(
          Qp + ((((size_t)b * N_HEADS + h0 + hh) * SEQ) + i0) * DIM_HEAD + kc * 32,
          DIM_HEAD, lane);

  v16h preA  = load_mixA_frag(pre, lane);
  v16h postA = load_mixA_frag(post, lane);

  int sk = tid >> 4, si = tid & 15;
  float mrun = -1e30f, lrun = 0.0f;
  v8f oacc[2][4] = {};
  int njt = (i0 + 31) / 32 + 1;

  for (int pass = 0; pass < 2; ++pass) {
    float inv_l = (pass == 1) ? (1.0f / lrun) : 0.0f;
    for (int jt = 0; jt < njt; ++jt) {
      int j0 = jt * 32;
      // --- step 1: per-head raw dots S = Q K^T (K transposed -> vectorized B frags) ---
#pragma unroll
      for (int hh = 0; hh < 2; ++hh) {
        int h = h0 + hh;
        const half_t* kthead = Kp + (((size_t)b * N_HEADS + h) * DIM_HEAD) * JP;
#pragma unroll
        for (int jh = 0; jh < 2; ++jh) {
          v8f s = {};
#pragma unroll
          for (int kc = 0; kc < 2; ++kc) {
            v16h kf = load_b_frag(kthead + (size_t)(kc * 32) * JP + j0 + jh * 16, JP, lane);
            s = wmma_f16(qf[hh][kc], kf, s);
          }
          int nlo = lane & 15, hi = (lane >> 4) * 8;
#pragma unroll
          for (int r = 0; r < 8; ++r)
            bufA[((hi + r) * 16 + h) * 32 + jh * 16 + nlo] = (half_t)s[r];
        }
      }
      __syncthreads();
      // --- step 2: pre-softmax head mix + causal mask -> bufB [k][i][j] (f32) ---
#pragma unroll
      for (int it = 0; it < 4; ++it) {
        int item = w * 4 + it;
        int i = item >> 1, jh = item & 1;
        v16h sb = load_mixB_frag(bufA + (size_t)(i * 16) * 32 + jh * 16, lane);
        v8f d = {};
        d = wmma_f16(preA, sb, d);
        int nlo = lane & 15, hi = (lane >> 4) * 8;
        int jg = j0 + jh * 16 + nlo, ig = i0 + i;
#pragma unroll
        for (int r = 0; r < 8; ++r) {
          float val = d[r];
          if (jg > ig + NMEM) val = -1e30f;
          bufB[((hi + r) * 16 + i) * 32 + jh * 16 + nlo] = val;
        }
      }
      __syncthreads();
      const float4* rv = (const float4*)(bufB + (size_t)(sk * 16 + si) * 32);
      if (pass == 0) {
        float vals[32], tmax = mrun;
#pragma unroll
        for (int q = 0; q < 8; ++q) {
          float4 t = rv[q];
          vals[4 * q] = t.x; vals[4 * q + 1] = t.y; vals[4 * q + 2] = t.z; vals[4 * q + 3] = t.w;
        }
#pragma unroll
        for (int j = 0; j < 32; ++j) tmax = fmaxf(tmax, vals[j]);
        float s = 0.0f;
#pragma unroll
        for (int j = 0; j < 32; ++j) s += __expf(vals[j] - tmax);
        lrun = lrun * __expf(mrun - tmax) + s;
        mrun = tmax;
        __syncthreads();
      } else {
        // --- normalize: P = exp(S'-m)/l -> bufA [i][k][j] (f16, vectorized) ---
        float vals[32];
#pragma unroll
        for (int q = 0; q < 8; ++q) {
          float4 t = rv[q];
          vals[4 * q] = t.x; vals[4 * q + 1] = t.y; vals[4 * q + 2] = t.z; vals[4 * q + 3] = t.w;
        }
        half_t* prow = bufA + (size_t)(si * 16 + sk) * 32;
#pragma unroll
        for (int c8 = 0; c8 < 4; ++c8) {
          v8h o;
#pragma unroll
          for (int t = 0; t < 8; ++t)
            o[t] = (half_t)(__expf(vals[c8 * 8 + t] - mrun) * inv_l);
          *(v8h*)(prow + c8 * 8) = o;
        }
        __syncthreads();
        // --- post-softmax head mix -> P' in bufBh [k2][i][j] (f16) ---
#pragma unroll
        for (int it = 0; it < 4; ++it) {
          int item = w * 4 + it;
          int i = item >> 1, jh = item & 1;
          v16h pb = load_mixB_frag(bufA + (size_t)(i * 16) * 32 + jh * 16, lane);
          v8f d = {};
          d = wmma_f16(postA, pb, d);
          int nlo = lane & 15, hi = (lane >> 4) * 8;
#pragma unroll
          for (int r = 0; r < 8; ++r)
            bufBh[((hi + r) * 16 + i) * 32 + jh * 16 + nlo] = (half_t)d[r];
        }
        __syncthreads();
        // --- O += P' V  (K = 32 = j-tile) ---
#pragma unroll
        for (int hh = 0; hh < 2; ++hh) {
          int h = h0 + hh;
          v16h pa = load_a_frag(bufBh + (size_t)h * 16 * 32, 32, lane);
          const half_t* vbase = Vp + ((((size_t)b * N_HEADS + h) * JP) + j0) * DIM_HEAD;
#pragma unroll
          for (int dt = 0; dt < 4; ++dt) {
            v16h vb = load_b_frag(vbase + dt * 16, DIM_HEAD, lane);
            oacc[hh][dt] = wmma_f16(pa, vb, oacc[hh][dt]);
          }
        }
        __syncthreads();
      }
    }
  }
  half_t* AO = ws + OFF_AO;
  int nlo = lane & 15, hi = (lane >> 4) * 8;
#pragma unroll
  for (int hh = 0; hh < 2; ++hh) {
    int h = h0 + hh;
#pragma unroll
    for (int dt = 0; dt < 4; ++dt)
#pragma unroll
      for (int r = 0; r < 8; ++r)
        AO[((size_t)(b * SEQ + i0 + hi + r)) * INNER + h * DIM_HEAD + dt * 16 + nlo] =
            (half_t)oacc[hh][dt][r];
  }
}

// ---------------- phase 3: output projection, f32 result ----------------
__global__ __launch_bounds__(256) void out_gemm_kernel(const half_t* __restrict__ ws,
                                                       float* __restrict__ out) {
  __shared__ half_t As[2][BM * LDA_S];
  __shared__ half_t Bs[2][BK * LDB_S];
  const half_t* Ag = ws + OFF_AO + (size_t)blockIdx.y * BM * INNER;
  const half_t* Bg = ws + OFF_WOH + blockIdx.x * BN;
  int tid = threadIdx.x, lane = tid & 31, w = tid >> 5;
  int m0 = blockIdx.y * BM;
  int c0 = blockIdx.x * BN;
  int wm = w >> 2, wn = w & 3;
  v8f acc[2][2] = {};

  stage_tile(As[0], Bs[0], Ag, INNER, Bg, D_MODEL, 0, tid);
  for (int t = 0; t < NT_QKV; ++t) {
    if (t + 1 < NT_QKV) {
      stage_tile(As[(t + 1) & 1], Bs[(t + 1) & 1], Ag, INNER, Bg, D_MODEL, (t + 1) * BK, tid);
      wait_async6();
    } else {
      wait_async0();
    }
    __syncthreads();
    const half_t* At = As[t & 1];
    const half_t* Bt = Bs[t & 1];
#pragma unroll
    for (int kc = 0; kc < 2; ++kc) {
      v16h af[2], bf[2];
#pragma unroll
      for (int f = 0; f < 2; ++f) {
        af[f] = load_a_frag(At + (wm * 32 + f * 16) * LDA_S + kc * 32, LDA_S, lane);
        bf[f] = load_b_frag(Bt + (size_t)(kc * 32) * LDB_S + wn * 32 + f * 16, LDB_S, lane);
      }
#pragma unroll
      for (int fm = 0; fm < 2; ++fm)
#pragma unroll
        for (int fn = 0; fn < 2; ++fn)
          acc[fm][fn] = wmma_f16(af[fm], bf[fn], acc[fm][fn]);
    }
    __syncthreads();
  }
  int nlo = lane & 15, hi = (lane >> 4) * 8;
#pragma unroll
  for (int fm = 0; fm < 2; ++fm)
#pragma unroll
    for (int fn = 0; fn < 2; ++fn)
#pragma unroll
      for (int r = 0; r < 8; ++r) {
        int mg = m0 + wm * 32 + fm * 16 + hi + r;
        int cg = c0 + wn * 32 + fn * 16 + nlo;
        out[(size_t)mg * D_MODEL + cg] = acc[fm][fn][r];
      }
}

extern "C" void kernel_launch(void* const* d_in, const int* in_sizes, int n_in,
                              void* d_out, int out_size, void* d_ws, size_t ws_size,
                              hipStream_t stream) {
  const float* x     = (const float*)d_in[0];
  const float* Wq    = (const float*)d_in[1];
  const float* Wkv   = (const float*)d_in[2];
  const float* Wo    = (const float*)d_in[3];
  const float* pre   = (const float*)d_in[4];
  const float* post  = (const float*)d_in[5];
  const float* mem_k = (const float*)d_in[6];
  const float* mem_v = (const float*)d_in[7];
  half_t* ws  = (half_t*)d_ws;
  float*  out = (float*)d_out;

  prep_kernel<<<4096, 256, 0, stream>>>(x, Wq, Wkv, Wo, mem_k, mem_v, ws);
  qkv_gemm_kernel<<<dim3(3 * INNER / BN, MROWS / BM), 256, 0, stream>>>(ws);
  attn_kernel<<<dim3(SEQ / 16, BATCH), 256, 0, stream>>>(pre, post, ws);
  out_gemm_kernel<<<dim3(D_MODEL / BN, MROWS / BM), 256, 0, stream>>>(ws, out);
}
